// WaveFieldAttention_69398081569385
// MI455X (gfx1250) — matrix operand order repaired and checked by reading, not compiled, not run on hardware
//
#include <hip/hip_runtime.h>
#include <math.h>

// ---------------------------------------------------------------------------
// WaveFieldAttention for MI455X (gfx1250, wave32).
// Heavy GEMMs use v_wmma_f32_16x16x32_bf16; FFT conv runs in LDS.
// ---------------------------------------------------------------------------

#define Bb 4
#define Nn 2048
#define Ee 1024
#define Hh 16
#define Dd 64
#define Gg 2048
#define FBb 2049
#define NCc 32

typedef __attribute__((ext_vector_type(8)))  float  v8f;
typedef __attribute__((ext_vector_type(8)))  __bf16 v8bf;
typedef __attribute__((ext_vector_type(16))) __bf16 v16bf;

// ------------------------- f32 -> bf16 conversion ---------------------------
__global__ void k_cvt_bf16(const float* __restrict__ s, __bf16* __restrict__ d, int n) {
    int i = blockIdx.x * blockDim.x + threadIdx.x;
    if (i < n) d[i] = (__bf16)s[i];
}

// ------------------------- WMMA bf16 GEMM -----------------------------------
// C[M,N] = A[M,K] * Bw[N,K]^T + bias[N].  M,N multiples of 128, K multiple of 32.
// Block: 256 threads = 8 waves (2 M-waves x 4 N-waves). Wave tile: 64x32.
__global__ __launch_bounds__(256)
void k_gemm_bf16(const __bf16* __restrict__ A, const __bf16* __restrict__ Bw,
                 const float* __restrict__ bias, float* __restrict__ C,
                 int M, int Ncols, int K) {
    const int lane  = threadIdx.x & 31;
    const int wave  = threadIdx.x >> 5;
    const int waveM = wave & 1;
    const int waveN = wave >> 1;
    const int mBase = blockIdx.y * 128 + waveM * 64;
    const int nBase = blockIdx.x * 128 + waveN * 32;
    const int half  = lane >> 4;      // 0: lanes 0-15, 1: lanes 16-31
    const int l15   = lane & 15;

    v8f acc[4][2];
    const v8f vzero = {0.f, 0.f, 0.f, 0.f, 0.f, 0.f, 0.f, 0.f};
#pragma unroll
    for (int i = 0; i < 4; ++i)
#pragma unroll
        for (int j = 0; j < 2; ++j) acc[i][j] = vzero;

    union U16 { v16bf v; struct { v8bf lo; v8bf hi; } s; };

    for (int k0 = 0; k0 < K; k0 += 32) {
        v16bf afr[4];
#pragma unroll
        for (int mt = 0; mt < 4; ++mt) {
            // 16-bit A 16x32 layout: lane half 0 -> K {0..7,16..23}, half 1 -> K {8..15,24..31}
            const __bf16* p = A + (size_t)(mBase + mt * 16 + l15) * K + k0 + half * 8;
            U16 u;
            u.s.lo = *reinterpret_cast<const v8bf*>(p);
            u.s.hi = *reinterpret_cast<const v8bf*>(p + 16);
            afr[mt] = u.v;
            __builtin_prefetch(p + 32, 0, 1);   // global_prefetch_b8 next K tile
        }
        v16bf bfr[2];
#pragma unroll
        for (int nt = 0; nt < 2; ++nt) {
            // B 32x16: lane = column, half 0 -> K {0..7,16..23}, half 1 -> K {8..15,24..31}
            const __bf16* p = Bw + (size_t)(nBase + nt * 16 + l15) * K + k0 + half * 8;
            U16 u;
            u.s.lo = *reinterpret_cast<const v8bf*>(p);
            u.s.hi = *reinterpret_cast<const v8bf*>(p + 16);
            bfr[nt] = u.v;
            __builtin_prefetch(p + 32, 0, 1);
        }
#pragma unroll
        for (int mt = 0; mt < 4; ++mt)
#pragma unroll
            for (int nt = 0; nt < 2; ++nt)
                acc[mt][nt] = __builtin_amdgcn_wmma_f32_16x16x32_bf16(
                    false, afr[mt], false, bfr[nt], (short)0, acc[mt][nt], false, false);
    }

    // C/D layout: VGPR r -> row r + 8*half, column = l15
#pragma unroll
    for (int nt = 0; nt < 2; ++nt) {
        const int col = nBase + nt * 16 + l15;
        const float bv = bias[col];
#pragma unroll
        for (int mt = 0; mt < 4; ++mt) {
#pragma unroll
            for (int r = 0; r < 8; ++r) {
                const int row = mBase + mt * 16 + r + half * 8;
                C[(size_t)row * Ncols + col] = acc[mt][nt][r] + bv;
            }
        }
    }
}

// ------------------------- q_feat / k_feat ----------------------------------
// qkvg layout [B*N, 4E]; q = [0,E), k = [E,2E).
// q_feat[tok, h*D+e] = relu(sum_d Wq[e,d]*q[tok,h*D+d] + bq[e]) + 1e-6
__global__ __launch_bounds__(256)
void k_feat(const float* __restrict__ qkvg,
            const float* __restrict__ Wq, const float* __restrict__ bq,
            const float* __restrict__ Wk, const float* __restrict__ bk,
            float* __restrict__ qfeat, float* __restrict__ kfeat) {
    __shared__ float sWq[Dd * Dd];
    __shared__ float sWk[Dd * Dd];
    __shared__ float sQ[Ee];
    __shared__ float sK[Ee];
    const int tok = blockIdx.x;
    const float* row = qkvg + (size_t)tok * (4 * Ee);
    for (int i = threadIdx.x; i < Dd * Dd; i += 256) { sWq[i] = Wq[i]; sWk[i] = Wk[i]; }
    for (int i = threadIdx.x; i < Ee; i += 256)      { sQ[i] = row[i]; sK[i] = row[Ee + i]; }
    __syncthreads();
    for (int o = threadIdx.x; o < Ee; o += 256) {
        const int h = o >> 6, e = o & 63;
        float aq = bq[e], ak = bk[e];
        const float* wq = sWq + e * Dd;
        const float* wk = sWk + e * Dd;
        const float* qh = sQ + h * Dd;
        const float* kh = sK + h * Dd;
#pragma unroll 8
        for (int d = 0; d < Dd; ++d) { aq += wq[d] * qh[d]; ak += wk[d] * kh[d]; }
        qfeat[(size_t)tok * Ee + o] = fmaxf(aq, 0.f) + 1e-6f;
        kfeat[(size_t)tok * Ee + o] = fmaxf(ak, 0.f) + 1e-6f;
    }
}

// ------------------- wave kernel + direct rDFT (base_fft) -------------------
__global__ __launch_bounds__(256)
void k_kernfft(const float* __restrict__ wf, const float* __restrict__ wd,
               const float* __restrict__ wp, float2* __restrict__ basefft) {
    __shared__ float skern[Gg];
    __shared__ float sred[256];
    const int h = blockIdx.x;
    const float alpha = log1pf(expf(wd[h]));   // softplus
    const float freq = wf[h], ph = wp[h];
    float part = 0.f;
    for (int t = threadIdx.x; t < Gg; t += 256) {
        float v = expf(-alpha * (float)t) * cosf(freq * (float)t + ph);
        skern[t] = v;
        part += fabsf(v);
    }
    sred[threadIdx.x] = part;
    __syncthreads();
    for (int s = 128; s > 0; s >>= 1) {
        if (threadIdx.x < s) sred[threadIdx.x] += sred[threadIdx.x + s];
        __syncthreads();
    }
    const float inv = 1.f / fmaxf(sred[0], 1e-8f);
    __syncthreads();
    for (int t = threadIdx.x; t < Gg; t += 256) skern[t] *= inv;
    __syncthreads();
    const float w0 = -6.2831853071795864f / (2.f * Gg);
    for (int f = threadIdx.x; f < FBb; f += 256) {
        float re = 0.f, im = 0.f;
        for (int t = 0; t < Gg; ++t) {
            float sn, cs;
            __sincosf(w0 * (float)f * (float)t, &sn, &cs);
            re += skern[t] * cs;
            im += skern[t] * sn;
        }
        basefft[h * FBb + f] = make_float2(re, im);
    }
}

// ------------------------- gate network -------------------------------------
__global__ void k_gate_ln(const float* __restrict__ qkvg, const float* __restrict__ ln_g,
                          const float* __restrict__ ln_b, float* __restrict__ qbar) {
    const int i = threadIdx.x;
    if (i >= Bb * Hh) return;
    const int b = i >> 4, h = i & 15;
    const float* q0 = qkvg + (size_t)b * Nn * (4 * Ee) + h * Dd;   // n = 0, q section
    float mu = 0.f;
    for (int d = 0; d < Dd; ++d) mu += q0[d];
    mu *= (1.f / Dd);
    float var = 0.f;
    for (int d = 0; d < Dd; ++d) { float t = q0[d] - mu; var += t * t; }
    var *= (1.f / Dd);
    const float is = rsqrtf(var + 1e-5f);
    for (int d = 0; d < Dd; ++d)
        qbar[b * Ee + h * Dd + d] = (q0[d] - mu) * is * ln_g[d] + ln_b[d];
}

__global__ void k_h1(const float* __restrict__ qbar, const float* __restrict__ W,
                     const float* __restrict__ bias, float* __restrict__ h1) {
    const int idx = blockIdx.x * blockDim.x + threadIdx.x;
    if (idx >= Bb * Ee) return;
    const int b = idx >> 10, e = idx & (Ee - 1);
    const float* wr = W + (size_t)e * Ee;
    const float* xr = qbar + b * Ee;
    float a = bias[e];
    for (int k = 0; k < Ee; ++k) a += wr[k] * xr[k];
    h1[idx] = 0.5f * a * (1.f + erff(a * 0.70710678118654752f));  // exact gelu
}

__global__ void k_ctrl(const float* __restrict__ h1, const float* __restrict__ W,
                       const float* __restrict__ bias, float* __restrict__ ctrl) {
    const int idx = blockIdx.x * blockDim.x + threadIdx.x;
    if (idx >= Bb * Hh * NCc) return;
    const int b = idx >> 9, j = idx & 511;
    const float* wr = W + (size_t)j * Ee;
    const float* xr = h1 + b * Ee;
    float a = bias[j];
    for (int k = 0; k < Ee; ++k) a += wr[k] * xr[k];
    ctrl[idx] = a;
}

__global__ void k_gateexp(const float* __restrict__ ctrl, float* __restrict__ gate) {
    const int idx = blockIdx.x * blockDim.x + threadIdx.x;
    if (idx >= Bb * Hh * FBb) return;
    const int b = idx / (Hh * FBb);
    const int r = idx % (Hh * FBb);
    const int h = r / FBb;
    const int f = r % FBb;
    const float src = (float)f * (float)(NCc - 1) / (float)(FBb - 1);
    int lo = (int)floorf(src);
    lo = min(max(lo, 0), NCc - 2);
    const float fr = src - (float)lo;
    const float* c = ctrl + (b * Hh + h) * NCc;
    gate[idx] = c[lo] * (1.f - fr) + c[lo + 1] * fr;
}

// ------------------------- scatter to field ---------------------------------
// field layout [B,H,2D,G] so each FFT sequence is contiguous.
__global__ void k_scatter(const float* __restrict__ qkvg, const float* __restrict__ kfeat,
                          float* __restrict__ field) {
    const long idx = (long)blockIdx.x * 256 + threadIdx.x;   // B*H*N*2D
    const int d2 = (int)(idx & 127);
    long r = idx >> 7;
    const int n = (int)(r & (Nn - 1)); r >>= 11;
    const int h = (int)(r & 15);
    const int b = (int)(r >> 4);
    const int d = d2 & (Dd - 1);
    const size_t tok = (size_t)(b * Nn + n);
    const float kf = kfeat[tok * Ee + h * Dd + d];
    const float val = (d2 < Dd) ? kf * qkvg[tok * (4 * Ee) + 2 * Ee + h * Dd + d] : kf;
    const float stride = (float)(Gg - 1) / (float)(Nn - 1);
    const float pos = (float)n * stride;
    int lo = (int)pos;
    lo = min(max(lo, 0), Gg - 2);
    const float fr = fminf(fmaxf(pos - (float)lo, 0.f), 1.f);
    float* fb = field + ((size_t)(b * Hh + h) * (2 * Dd) + d2) * Gg;
    atomicAdd(fb + lo,     val * (1.f - fr));
    atomicAdd(fb + lo + 1, val * fr);
}

// ------------------------- FFT convolution ----------------------------------
// Stockham radix-2, length 4096, ping-pong in LDS. 12 stages -> result back in src.
__device__ inline void fft4096(float2*& src, float2*& dst, float sign) {
    int st = 0;
    for (int ncur = 4096; ncur > 1; ncur >>= 1, ++st) {
        const int m = ncur >> 1;
        const float theta = sign * 6.2831853071795864f / (float)ncur;
        const int s = 1 << st;
        for (int idx = threadIdx.x; idx < 2048; idx += 256) {
            const int p = idx >> st;
            const int q = idx & (s - 1);
            float sn, cs;
            __sincosf(theta * (float)p, &sn, &cs);
            const float2 a = src[q + (p << st)];
            const float2 b = src[q + ((p + m) << st)];
            dst[q + ((2 * p) << st)] = make_float2(a.x + b.x, a.y + b.y);
            const float dx = a.x - b.x, dy = a.y - b.y;
            dst[q + ((2 * p + 1) << st)] = make_float2(dx * cs - dy * sn, dx * sn + dy * cs);
        }
        __syncthreads();
        float2* t = src; src = dst; dst = t;
    }
}

// One block per (b,h,d2) sequence; conv written back in place into `field`.
__global__ __launch_bounds__(256)
void k_fftconv(float* __restrict__ field, const float2* __restrict__ basefft,
               const float* __restrict__ gate) {
    __shared__ float2 bufA[4096];
    __shared__ float2 bufB[4096];
    const int gid = blockIdx.x;                 // (b*H + h)*2D + d2
    const int bh = gid >> 7;
    const int h = bh & 15;
    const int b = bh >> 4;
    float* seq = field + (size_t)gid * Gg;
    for (int t = threadIdx.x; t < 4096; t += 256)
        bufA[t] = make_float2(t < Gg ? seq[t] : 0.f, 0.f);
    __syncthreads();
    float2* src = bufA;
    float2* dst = bufB;
    fft4096(src, dst, -1.f);                    // forward; result in src (== bufA)
    for (int j = threadIdx.x; j < 4096; j += 256) {
        const int f = (j <= Gg) ? j : (4096 - j);
        const float2 bf = basefft[h * FBb + f];
        const float gv = 1.f + gate[(b * Hh + h) * FBb + f];
        const float kr = bf.x * gv;
        const float ki = ((j <= Gg) ? bf.y : -bf.y) * gv;
        const float2 v = src[j];
        src[j] = make_float2(v.x * kr - v.y * ki, v.x * ki + v.y * kr);
    }
    __syncthreads();
    fft4096(src, dst, 1.f);                     // inverse (unscaled)
    const float scale = 1.f / 4096.f;
    for (int t = threadIdx.x; t < Gg; t += 256) seq[t] = src[t].x * scale;
}

// ------------------------- H x H coupling -----------------------------------
__global__ __launch_bounds__(256)
void k_couple(const float* __restrict__ fc, const float* __restrict__ conv,
              float* __restrict__ convm) {
    __shared__ float sfc[Hh * Hh];
    if (threadIdx.x < Hh * Hh) sfc[threadIdx.x] = fc[threadIdx.x];
    __syncthreads();
    const long idx = (long)blockIdx.x * 256 + threadIdx.x;    // B*2D*G
    const int g = (int)(idx & (Gg - 1));
    long r = idx >> 11;
    const int d2 = (int)(r & 127);
    const int b = (int)(r >> 7);
    float in[Hh];
#pragma unroll
    for (int j = 0; j < Hh; ++j)
        in[j] = conv[((size_t)(b * Hh + j) * (2 * Dd) + d2) * Gg + g];
#pragma unroll
    for (int i = 0; i < Hh; ++i) {
        float a = 0.f;
#pragma unroll
        for (int j = 0; j < Hh; ++j) a += sfc[i * Hh + j] * in[j];
        convm[((size_t)(b * Hh + i) * (2 * Dd) + d2) * Gg + g] = a;
    }
}

// ------------------------- gather / attention / gating ----------------------
__global__ void k_attn(const float* __restrict__ qkvg, const float* __restrict__ qfeat,
                       const float* __restrict__ convm, __bf16* __restrict__ ybf) {
    const long idx = (long)blockIdx.x * 256 + threadIdx.x;    // B*N*E
    const int e = (int)(idx & (Ee - 1));
    long r = idx >> 10;
    const int n = (int)(r & (Nn - 1));
    const int b = (int)(r >> 11);
    const int h = e >> 6, d = e & (Dd - 1);
    const float stride = (float)(Gg - 1) / (float)(Nn - 1);
    const float pos = (float)n * stride;
    int lo = (int)pos;
    lo = min(max(lo, 0), Gg - 2);
    const float fr = fminf(fmaxf(pos - (float)lo, 0.f), 1.f);
    const float wlo = 1.f - fr, whi = fr;
    const float* cmN = convm + ((size_t)(b * Hh + h) * (2 * Dd) + d) * Gg;
    const float* cmD = convm + ((size_t)(b * Hh + h) * (2 * Dd) + Dd + d) * Gg;
    const float num = cmN[lo] * wlo + cmN[lo + 1] * whi;
    const float den = cmD[lo] * wlo + cmD[lo + 1] * whi;
    const float qf = qfeat[idx];
    const float attn = qf * num / (qf * den + 1e-6f);
    const float g = qkvg[(size_t)(b * Nn + n) * (4 * Ee) + 3 * Ee + e];
    const float y = attn / (1.f + expf(-g));                  // attn * sigmoid(g)
    ybf[idx] = (__bf16)y;
}

// ---------------------------------------------------------------------------
extern "C" void kernel_launch(void* const* d_in, const int* in_sizes, int n_in,
                              void* d_out, int out_size, void* d_ws, size_t ws_size,
                              hipStream_t stream) {
    const float* x       = (const float*)d_in[0];
    const float* W_qkvg  = (const float*)d_in[1];
    const float* b_qkvg  = (const float*)d_in[2];
    const float* W_out   = (const float*)d_in[3];
    const float* b_out   = (const float*)d_in[4];
    const float* W_qfm   = (const float*)d_in[5];
    const float* b_qfm   = (const float*)d_in[6];
    const float* W_kfm   = (const float*)d_in[7];
    const float* b_kfm   = (const float*)d_in[8];
    const float* ln_g    = (const float*)d_in[9];
    const float* ln_b    = (const float*)d_in[10];
    const float* W_g1    = (const float*)d_in[11];
    const float* b_g1    = (const float*)d_in[12];
    const float* W_g2    = (const float*)d_in[13];
    const float* b_g2    = (const float*)d_in[14];
    const float* wfreq   = (const float*)d_in[15];
    const float* wdamp   = (const float*)d_in[16];
    const float* wphase  = (const float*)d_in[17];
    const float* fcoup   = (const float*)d_in[18];
    float* out = (float*)d_out;

    char* w = (char*)d_ws;
    auto alloc = [&](size_t bytes) -> void* {
        void* p = (void*)w;
        w += (bytes + 255) & ~(size_t)255;
        return p;
    };
    const size_t nTok = (size_t)Bb * Nn;                 // 8192
    __bf16* xb     = (__bf16*)alloc(nTok * Ee * 2);
    __bf16* wqkvgb = (__bf16*)alloc((size_t)4 * Ee * Ee * 2);
    __bf16* woutb  = (__bf16*)alloc((size_t)Ee * Ee * 2);
    float*  qkvg   = (float*) alloc(nTok * 4 * Ee * 4);
    float*  qfeat  = (float*) alloc(nTok * Ee * 4);
    float*  kfeat  = (float*) alloc(nTok * Ee * 4);
    float2* bfft   = (float2*)alloc((size_t)Hh * FBb * 8);
    float*  qbar   = (float*) alloc((size_t)Bb * Ee * 4);
    float*  h1buf  = (float*) alloc((size_t)Bb * Ee * 4);
    float*  ctrlb  = (float*) alloc((size_t)Bb * Hh * NCc * 4);
    float*  gateb  = (float*) alloc((size_t)Bb * Hh * FBb * 4);
    const size_t fieldElems = (size_t)Bb * Hh * 2 * Dd * Gg;   // 16.78M
    float*  field  = (float*) alloc(fieldElems * 4);           // reused as conv
    float*  convm  = (float*) alloc(fieldElems * 4);
    __bf16* ybf    = (__bf16*)alloc(nTok * Ee * 2);

    // 1) conversions to bf16
    {
        int n1 = (int)(nTok * Ee);
        k_cvt_bf16<<<(n1 + 255) / 256, 256, 0, stream>>>(x, xb, n1);
        int n2 = 4 * Ee * Ee;
        k_cvt_bf16<<<(n2 + 255) / 256, 256, 0, stream>>>(W_qkvg, wqkvgb, n2);
        int n3 = Ee * Ee;
        k_cvt_bf16<<<(n3 + 255) / 256, 256, 0, stream>>>(W_out, woutb, n3);
    }
    // 2) qkvg GEMM: [8192,1024] x [4096,1024]^T (WMMA)
    k_gemm_bf16<<<dim3(4 * Ee / 128, (int)(nTok / 128)), 256, 0, stream>>>(
        xb, wqkvgb, b_qkvg, qkvg, (int)nTok, 4 * Ee, Ee);
    // 3) feature maps
    k_feat<<<(int)nTok, 256, 0, stream>>>(qkvg, W_qfm, b_qfm, W_kfm, b_kfm, qfeat, kfeat);
    // 4) base_fft of wave kernel
    k_kernfft<<<Hh, 256, 0, stream>>>(wfreq, wdamp, wphase, bfft);
    // 5) gate MLP
    k_gate_ln<<<1, 64, 0, stream>>>(qkvg, ln_g, ln_b, qbar);
    k_h1<<<(Bb * Ee + 255) / 256, 256, 0, stream>>>(qbar, W_g1, b_g1, h1buf);
    k_ctrl<<<(Bb * Hh * NCc + 255) / 256, 256, 0, stream>>>(h1buf, W_g2, b_g2, ctrlb);
    k_gateexp<<<(Bb * Hh * FBb + 255) / 256, 256, 0, stream>>>(ctrlb, gateb);
    // 6) field scatter
    hipMemsetAsync(field, 0, fieldElems * 4, stream);
    {
        long nsc = (long)Bb * Hh * Nn * 2 * Dd;          // 16,777,216
        k_scatter<<<(int)(nsc / 256), 256, 0, stream>>>(qkvg, kfeat, field);
    }
    // 7) FFT convolution (in place into field)
    k_fftconv<<<(int)(Bb * Hh * 2 * Dd), 256, 0, stream>>>(field, bfft, gateb);
    // 8) head coupling
    k_couple<<<(int)((size_t)Bb * 2 * Dd * Gg / 256), 256, 0, stream>>>(fcoup, field, convm);
    // 9) gather + attention + gating -> bf16 y
    k_attn<<<(int)(nTok * Ee / 256), 256, 0, stream>>>(qkvg, qfeat, convm, ybf);
    // 10) output GEMM: [8192,1024] x [1024,1024]^T (WMMA)
    k_gemm_bf16<<<dim3(Ee / 128, (int)(nTok / 128)), 256, 0, stream>>>(
        ybf, woutb, b_out, out, (int)nTok, Ee, Ee);

    (void)in_sizes; (void)n_in; (void)out_size; (void)ws_size;
}
